// TransformerBlock_20005957665361
// MI455X (gfx1250) — compile-verified, matrix operand order checked by reading, not run on hardware
//
#include <hip/hip_runtime.h>

// ---------------------------------------------------------------------------
// CDNA5 (gfx1250) transformer block, bf16 WMMA everywhere, wave32.
//   v_wmma_f32_16x16x32_bf16: D(16x16 f32) = A(16x32 bf16) * B(32x16 bf16) + C
// Fragment layouts (ISA 7.12.2, 16-bit operands, wave32):
//   A: lane L holds row M=L%16; elements 0..7 -> K = o..o+7, 8..15 -> K = 16+o..16+o+7
//      where o = (L<16) ? 0 : 8.   (two contiguous 16B runs per lane)
//   B: lane L holds col N=L%16 with the same K striping (so B is naturally fed
//      from a row-major [N,K] "transposed weight" array).
//   C/D: lane L holds col N=L%16; element i -> row M = i + (L<16 ? 0 : 8).
// ---------------------------------------------------------------------------

typedef __attribute__((ext_vector_type(16))) __bf16 v16bf;
typedef __attribute__((ext_vector_type(8)))  float  v8f;

#define WMMA_BF16(a, b, c) \
  __builtin_amdgcn_wmma_f32_16x16x32_bf16(false, (a), false, (b), (short)0, (c), false, false)

// Load one lane's 16 bf16 fragment elements: 2 x 16-byte chunks at p and p+16.
// Caller passes p = row_base + k0 + o  (o = lane<16 ? 0 : 8).
__device__ __forceinline__ v16bf load_frag(const __bf16* p) {
  v16bf r;
  *((uint4*)&r)     = *((const uint4*)p);
  *((uint4*)&r + 1) = *((const uint4*)(p + 16));
  return r;
}

// ---------------------------------------------------------------------------
// Generic GEMM: out[M,N] = act(A[M,K] @ BT[N,K]^T + bias) (+resid).
// One wave per 32x64 tile.  Ping-pong double buffering over a 64-wide k-step:
// two named register sets (A/B) alternate as load target / WMMA source, so
// there are NO rotate movs (and therefore no WMMA->VALU hazard NOPs), while
// the other set's 12 global_load_b128s stay in flight under the 8 WMMAs.
// Requires K % 64 == 0 (true here: K = 1024 or 4096).
// ---------------------------------------------------------------------------
__global__ __launch_bounds__(32)
void gemm_wmma_kernel(const __bf16* __restrict__ A, const __bf16* __restrict__ BT,
                      const float* __restrict__ bias, const float* __restrict__ resid,
                      float* __restrict__ outF, __bf16* __restrict__ outB,
                      int M, int N, int K, int relu)
{
  const int lane  = threadIdx.x;
  const int o     = (lane & 16) ? 8 : 0;
  const int r     = lane & 15;
  const int rbase = (lane & 16) ? 8 : 0;
  const size_t m0 = (size_t)blockIdx.x * 32;
  const size_t n0 = (size_t)blockIdx.y * 64;

  const __bf16* arow0 = A + (m0 + r) * (size_t)K + o;
  const __bf16* arow1 = A + (m0 + 16 + r) * (size_t)K + o;
  const __bf16* brow0 = BT + (n0 + r)      * (size_t)K + o;
  const __bf16* brow1 = BT + (n0 + 16 + r) * (size_t)K + o;
  const __bf16* brow2 = BT + (n0 + 32 + r) * (size_t)K + o;
  const __bf16* brow3 = BT + (n0 + 48 + r) * (size_t)K + o;

  v8f acc[2][4];
  #pragma unroll
  for (int mh = 0; mh < 2; ++mh)
    #pragma unroll
    for (int j = 0; j < 4; ++j) acc[mh][j] = (v8f)0.0f;

  v16bf a0A, a1A, b0A, b1A, b2A, b3A;
  v16bf a0B, a1B, b0B, b1B, b2B, b3B;

#define GEMM_LOADSET(S, kk)              \
  a0##S = load_frag(arow0 + (kk));       \
  a1##S = load_frag(arow1 + (kk));       \
  b0##S = load_frag(brow0 + (kk));       \
  b1##S = load_frag(brow1 + (kk));       \
  b2##S = load_frag(brow2 + (kk));       \
  b3##S = load_frag(brow3 + (kk));

#define GEMM_MMASET(S)                                \
  acc[0][0] = WMMA_BF16(a0##S, b0##S, acc[0][0]);     \
  acc[0][1] = WMMA_BF16(a0##S, b1##S, acc[0][1]);     \
  acc[0][2] = WMMA_BF16(a0##S, b2##S, acc[0][2]);     \
  acc[0][3] = WMMA_BF16(a0##S, b3##S, acc[0][3]);     \
  acc[1][0] = WMMA_BF16(a1##S, b0##S, acc[1][0]);     \
  acc[1][1] = WMMA_BF16(a1##S, b1##S, acc[1][1]);     \
  acc[1][2] = WMMA_BF16(a1##S, b2##S, acc[1][2]);     \
  acc[1][3] = WMMA_BF16(a1##S, b3##S, acc[1][3]);

  GEMM_LOADSET(A, 0)
  for (int k = 0; k < K; k += 64) {
    GEMM_LOADSET(B, k + 32)      // other set's loads fly under set-A WMMAs
    GEMM_MMASET(A)
    if (k + 64 < K) { GEMM_LOADSET(A, k + 64) }
    GEMM_MMASET(B)
  }
#undef GEMM_LOADSET
#undef GEMM_MMASET

  #pragma unroll
  for (int mh = 0; mh < 2; ++mh) {
    #pragma unroll
    for (int j = 0; j < 4; ++j) {
      #pragma unroll
      for (int i = 0; i < 8; ++i) {
        size_t mm = m0 + (size_t)mh * 16 + rbase + i;
        size_t nn = n0 + (size_t)j * 16 + r;
        float v = acc[mh][j][i];
        if (bias)  v += bias[nn];
        if (relu)  v = fmaxf(v, 0.0f);
        if (resid) v += resid[mm * (size_t)N + nn];
        if (outF)  outF[mm * (size_t)N + nn] = v;
        if (outB)  outB[mm * (size_t)N + nn] = (__bf16)v;
      }
    }
  }
}

// ---------------------------------------------------------------------------
// Fused causal flash attention, one wave per (b, h, 16-query tile).
// Q,K,V are bf16 [B*T, C] with head h occupying cols [h*HS, h*HS+64).
// scale = C^-0.5 = 1/32 (the reference's quirky full-dim scaling).
// ---------------------------------------------------------------------------
__global__ __launch_bounds__(32)
void attn_wmma_kernel(const __bf16* __restrict__ Q, const __bf16* __restrict__ Kt,
                      const __bf16* __restrict__ V, __bf16* __restrict__ O,
                      int T, int C, int H, int HS)
{
  __shared__ __align__(16) __bf16 VT[64][40];  // V^T tile: [d][s], padded stride 40
  __shared__ __align__(16) __bf16 PS[16][40];  // P tile (D-layout -> A-layout bounce)

  const int lane  = threadIdx.x;
  const int o     = (lane & 16) ? 8 : 0;
  const int r     = lane & 15;
  const int rbase = (lane & 16) ? 8 : 0;
  const int bh = blockIdx.x;
  const int b  = bh / H, h = bh % H;
  const int t0 = blockIdx.y * 16;

  const __bf16* qb = Q  + ((size_t)b * T) * C + (size_t)h * HS;
  const __bf16* kb = Kt + ((size_t)b * T) * C + (size_t)h * HS;
  const __bf16* vb = V  + ((size_t)b * T) * C + (size_t)h * HS;

  // Q A-fragments for this query tile (K-dim = HS = 64 -> two k-steps).
  v16bf qa0 = load_frag(qb + (size_t)(t0 + r) * C + 0  + o);
  v16bf qa1 = load_frag(qb + (size_t)(t0 + r) * C + 32 + o);

  v8f Oacc[4];
  #pragma unroll
  for (int j = 0; j < 4; ++j) Oacc[j] = (v8f)0.0f;
  float mrow[8], lrow[8];
  #pragma unroll
  for (int i = 0; i < 8; ++i) { mrow[i] = -__builtin_inff(); lrow[i] = 0.0f; }
  const float scale = 0.03125f;  // 1024^-0.5

  for (int s0 = 0; s0 < t0 + 16; s0 += 32) {
    // Stage V^T [64 d][32 s] in LDS (lane -> one key row).
    __syncthreads();
    {
      const __bf16* vrow = vb + (size_t)(s0 + lane) * C;
      #pragma unroll
      for (int d = 0; d < 64; ++d) VT[d][lane] = vrow[d];
    }

    // Scores S[16 t][32 s] = Q @ K^T, D-layout (lane = key col, elem = query row).
    float sv[2][8];
    #pragma unroll
    for (int sub = 0; sub < 2; ++sub) {
      int srow = s0 + sub * 16 + r;
      v16bf kf0 = load_frag(kb + (size_t)srow * C + 0  + o);
      v16bf kf1 = load_frag(kb + (size_t)srow * C + 32 + o);
      v8f s = (v8f)0.0f;
      s = WMMA_BF16(qa0, kf0, s);
      s = WMMA_BF16(qa1, kf1, s);
      #pragma unroll
      for (int i = 0; i < 8; ++i) {
        int t = t0 + rbase + i;
        float v = s[i] * scale;
        if (srow > t) v = -__builtin_inff();  // causal mask
        sv[sub][i] = v;
      }
    }

    // Online softmax: row reductions across the 16-lane half (xor 1,2,4,8).
    float mnew[8], alpha[8];
    #pragma unroll
    for (int i = 0; i < 8; ++i) {
      float m = fmaxf(sv[0][i], sv[1][i]);
      #pragma unroll
      for (int msk = 1; msk < 16; msk <<= 1) m = fmaxf(m, __shfl_xor(m, msk, 32));
      mnew[i]  = fmaxf(mrow[i], m);
      alpha[i] = __expf(mrow[i] - mnew[i]);
      mrow[i]  = mnew[i];
    }
    #pragma unroll
    for (int sub = 0; sub < 2; ++sub)
      #pragma unroll
      for (int i = 0; i < 8; ++i)
        sv[sub][i] = __expf(sv[sub][i] - mnew[i]);
    #pragma unroll
    for (int i = 0; i < 8; ++i) {
      float rs = sv[0][i] + sv[1][i];
      #pragma unroll
      for (int msk = 1; msk < 16; msk <<= 1) rs += __shfl_xor(rs, msk, 32);
      lrow[i] = lrow[i] * alpha[i] + rs;
    }
    #pragma unroll
    for (int j = 0; j < 4; ++j)
      #pragma unroll
      for (int i = 0; i < 8; ++i) Oacc[j][i] *= alpha[i];

    // D-layout -> A-layout via LDS (same wave; DS ops are in-order).
    #pragma unroll
    for (int sub = 0; sub < 2; ++sub)
      #pragma unroll
      for (int i = 0; i < 8; ++i)
        PS[rbase + i][sub * 16 + r] = (__bf16)sv[sub][i];
    __syncthreads();

    v16bf pa = load_frag((const __bf16*)&PS[r][o]);
    #pragma unroll
    for (int j = 0; j < 4; ++j) {
      v16bf bv = load_frag((const __bf16*)&VT[j * 16 + r][o]);
      Oacc[j] = WMMA_BF16(pa, bv, Oacc[j]);
    }
  }

  // Normalize and scatter (heads are interleaved back to [B*T, C]).
  __bf16* ob = O + ((size_t)b * T) * C + (size_t)h * HS;
  #pragma unroll
  for (int j = 0; j < 4; ++j)
    #pragma unroll
    for (int i = 0; i < 8; ++i) {
      int t = t0 + rbase + i;
      ob[(size_t)t * C + j * 16 + r] = (__bf16)(Oacc[j][i] / lrow[i]);
    }
}

// ---------------------------------------------------------------------------
// LayerNorm over C per row; emits f32 (residual base) and bf16 (GEMM operand).
// ---------------------------------------------------------------------------
__global__ __launch_bounds__(256)
void layernorm_kernel(const float* __restrict__ x, const float* __restrict__ g,
                      const float* __restrict__ be, float* __restrict__ outF,
                      __bf16* __restrict__ outB, int C)
{
  const int row = blockIdx.x;
  const int tid = threadIdx.x;
  const float* xr = x + (size_t)row * C;

  float s = 0.0f, ss = 0.0f;
  for (int i = tid; i < C; i += 256) { float v = xr[i]; s += v; ss += v * v; }
  #pragma unroll
  for (int off = 16; off > 0; off >>= 1) {
    s  += __shfl_down(s,  off, 32);
    ss += __shfl_down(ss, off, 32);
  }
  __shared__ float as[8], bs[8];
  __shared__ float mu_s, rstd_s;
  int w = tid >> 5, ln = tid & 31;
  if (ln == 0) { as[w] = s; bs[w] = ss; }
  __syncthreads();
  if (tid == 0) {
    float S = 0.0f, SS = 0.0f;
    #pragma unroll
    for (int i = 0; i < 8; ++i) { S += as[i]; SS += bs[i]; }
    float mu  = S / C;
    float var = SS / C - mu * mu;
    mu_s = mu; rstd_s = rsqrtf(var + 1e-5f);
  }
  __syncthreads();
  float mu = mu_s, rstd = rstd_s;
  for (int i = tid; i < C; i += 256) {
    float v = (xr[i] - mu) * rstd * g[i] + be[i];
    if (outF) outF[(size_t)row * C + i] = v;
    outB[(size_t)row * C + i] = (__bf16)v;
  }
}

// ---------------------------------------------------------------------------
// Weight prep: fp32 -> bf16, transposed to row-major [N, K].
// ---------------------------------------------------------------------------
__global__ void prep_transpose_kernel(const float* __restrict__ W, __bf16* __restrict__ WT,
                                      int K, int N)
{
  size_t e = (size_t)blockIdx.x * blockDim.x + threadIdx.x;
  size_t total = (size_t)K * N;
  if (e >= total) return;
  size_t k = e / N, n = e % N;
  WT[n * (size_t)K + k] = (__bf16)W[e];
}

// Wqkv: (H, C, HS) -> bf16 [C_out = h*HS+d][C], so concatenated heads form one GEMM.
__global__ void prep_qkvw_kernel(const float* __restrict__ W, __bf16* __restrict__ WT,
                                 int C, int HS)
{
  size_t e = (size_t)blockIdx.x * blockDim.x + threadIdx.x;
  size_t total = (size_t)C * C;
  if (e >= total) return;
  size_t h   = e / ((size_t)C * HS);
  size_t rem = e % ((size_t)C * HS);
  size_t c = rem / HS, d = rem % HS;
  WT[(h * HS + d) * (size_t)C + c] = (__bf16)W[e];
}

// ---------------------------------------------------------------------------
extern "C" void kernel_launch(void* const* d_in, const int* in_sizes, int n_in,
                              void* d_out, int out_size, void* d_ws, size_t ws_size,
                              hipStream_t stream)
{
  (void)in_sizes; (void)n_in; (void)out_size; (void)ws_size;
  constexpr int B = 2, T = 2048, C = 1024, H = 16, HS = 64;
  constexpr int M = B * T;        // 4096 rows
  constexpr int F = 4 * C;        // 4096 hidden

  const float* x   = (const float*)d_in[0];
  const float* Wq  = (const float*)d_in[1];
  const float* Wk  = (const float*)d_in[2];
  const float* Wv  = (const float*)d_in[3];
  const float* Wp  = (const float*)d_in[4];
  const float* bp  = (const float*)d_in[5];
  const float* W1  = (const float*)d_in[6];
  const float* b1  = (const float*)d_in[7];
  const float* W2  = (const float*)d_in[8];
  const float* b2  = (const float*)d_in[9];
  const float* g1  = (const float*)d_in[10];
  const float* be1 = (const float*)d_in[11];
  const float* g2  = (const float*)d_in[12];
  const float* be2 = (const float*)d_in[13];

  // Bump-allocate workspace (256B aligned). ~152 MB total.
  char* wsp = (char*)d_ws;
  auto alloc = [&](size_t bytes) -> void* {
    void* p = (void*)wsp;
    wsp += (bytes + 255) & ~(size_t)255;
    return p;
  };
  float*  xln_f = (float*) alloc((size_t)M * C * 4);
  float*  y_f   = (float*) alloc((size_t)M * C * 4);
  float*  h_f   = (float*) alloc((size_t)M * C * 4);
  __bf16* xln_b = (__bf16*)alloc((size_t)M * C * 2);
  __bf16* h_b   = (__bf16*)alloc((size_t)M * C * 2);
  __bf16* att_b = (__bf16*)alloc((size_t)M * C * 2);
  __bf16* Qb    = (__bf16*)alloc((size_t)M * C * 2);
  __bf16* Kb    = (__bf16*)alloc((size_t)M * C * 2);
  __bf16* Vb    = (__bf16*)alloc((size_t)M * C * 2);
  __bf16* WqT   = (__bf16*)alloc((size_t)C * C * 2);
  __bf16* WkT   = (__bf16*)alloc((size_t)C * C * 2);
  __bf16* WvT   = (__bf16*)alloc((size_t)C * C * 2);
  __bf16* WpT   = (__bf16*)alloc((size_t)C * C * 2);
  __bf16* W1T   = (__bf16*)alloc((size_t)C * F * 2);
  __bf16* W2T   = (__bf16*)alloc((size_t)F * C * 2);
  __bf16* ff1_b = (__bf16*)alloc((size_t)M * F * 2);

  // 1) Weight prep
  {
    int g_cc = (C * C + 255) / 256;
    int g_cf = (C * F + 255) / 256;
    prep_qkvw_kernel<<<g_cc, 256, 0, stream>>>(Wq, WqT, C, HS);
    prep_qkvw_kernel<<<g_cc, 256, 0, stream>>>(Wk, WkT, C, HS);
    prep_qkvw_kernel<<<g_cc, 256, 0, stream>>>(Wv, WvT, C, HS);
    prep_transpose_kernel<<<g_cc, 256, 0, stream>>>(Wp, WpT, C, C);
    prep_transpose_kernel<<<g_cf, 256, 0, stream>>>(W1, W1T, C, F);
    prep_transpose_kernel<<<g_cf, 256, 0, stream>>>(W2, W2T, F, C);
  }

  // 2) LN1: x -> xln (f32 residual base + bf16 operand)
  layernorm_kernel<<<M, 256, 0, stream>>>(x, g1, be1, xln_f, xln_b, C);

  // 3) QKV projections (bf16 WMMA), outputs bf16
  {
    dim3 grid(M / 32, C / 64);
    gemm_wmma_kernel<<<grid, 32, 0, stream>>>(xln_b, WqT, nullptr, nullptr, nullptr, Qb, M, C, C, 0);
    gemm_wmma_kernel<<<grid, 32, 0, stream>>>(xln_b, WkT, nullptr, nullptr, nullptr, Kb, M, C, C, 0);
    gemm_wmma_kernel<<<grid, 32, 0, stream>>>(xln_b, WvT, nullptr, nullptr, nullptr, Vb, M, C, C, 0);
  }

  // 4) Fused causal flash attention
  {
    dim3 grid(B * H, T / 16);
    attn_wmma_kernel<<<grid, 32, 0, stream>>>(Qb, Kb, Vb, att_b, T, C, H, HS);
  }

  // 5) Output projection + bias + residual(xln) -> y
  {
    dim3 grid(M / 32, C / 64);
    gemm_wmma_kernel<<<grid, 32, 0, stream>>>(att_b, WpT, bp, xln_f, y_f, nullptr, M, C, C, 0);
  }

  // 6) LN2: y -> h (f32 residual base + bf16 operand)
  layernorm_kernel<<<M, 256, 0, stream>>>(y_f, g2, be2, h_f, h_b, C);

  // 7) FFN1: relu(h @ W1 + b1) -> bf16
  {
    dim3 grid(M / 32, F / 64);
    gemm_wmma_kernel<<<grid, 32, 0, stream>>>(h_b, W1T, b1, nullptr, nullptr, ff1_b, M, F, C, 1);
  }

  // 8) FFN2: ff1 @ W2 + b2 + h -> d_out (fp32)
  {
    dim3 grid(M / 32, C / 64);
    gemm_wmma_kernel<<<grid, 32, 0, stream>>>(ff1_b, W2T, b2, h_f, (float*)d_out, nullptr, M, C, F, 0);
  }
}
